// RNNClassifier_76338748719515
// MI455X (gfx1250) — matrix operand order repaired and checked by reading, not compiled, not run on hardware
//
#include <hip/hip_runtime.h>

// CTRNN fused kernel for MI455X (gfx1250, wave32, WMMA).
// B=256, T=1024, F=64, H=128, O=10, 6 Euler substeps, DT=0.1, TAU=1.
//
// Recurrence is batch-row independent -> 16 workgroups (16 batch rows each),
// no global sync. 8 waves per WG, wave w owns hidden columns [16w,16w+16).
// h/xp/acc live in v8f registers in the WMMA f32 C layout; a double-buffered
// shared f16 h-tile is used only to re-gather the A matrix each substep,
// needing a single barrier per substep.

typedef __attribute__((ext_vector_type(16))) _Float16 v16h;
typedef __attribute__((ext_vector_type(8)))  _Float16 v8h;
typedef __attribute__((ext_vector_type(8)))  float    v8f;
typedef __attribute__((ext_vector_type(4)))  float    v4f;

#define B_      256
#define T_      1024
#define F_      64
#define H_      128
#define O_      10
#define UNFOLD_ 6
#define DT_     0.1f

#define LDW 136   // f16 row stride for weight stage + h buffers (16B aligned)
#define LDX 72    // f16 row stride for x tile (16B aligned)

// A-matrix (16xK f16) fragment gather for one K-step of 32.
// Layout: lanes 0-15 = row M=lane, K in k0..k0+15;
//         lanes 16-31 = row M=lane-16, K in k0+16..k0+31.
__device__ __forceinline__ v16h load_afrag(const _Float16* buf, int ldp,
                                           int lane, int k0) {
    const int half = lane >> 4;
    const int row  = lane & 15;
    const int ka   = k0 + (half ? 16 : 0);
    const v8h lo = *(const v8h*)(buf + row * ldp + ka);
    const v8h hi = *(const v8h*)(buf + row * ldp + ka + 8);
    v16h a;
#pragma unroll
    for (int i = 0; i < 8; ++i) { a[i] = lo[i]; a[8 + i] = hi[i]; }
    return a;
}

// B-matrix (32x16 f16) fragment gather for one K-step of 32, column base c0.
// Contiguous-k layout (same as load_afrag): lanes 0-15 hold K=k0..k0+15 (2 per VGPR), lanes 16-31 K=k0+16..k0+31;
// N = lane & 15.
__device__ __forceinline__ v16h load_bfrag(const _Float16* buf, int ldp,
                                           int lane, int k0, int c0) {
    const int n    = lane & 15;
    const int koff = (lane >> 4) ? 16 : 0;
    v16h b;
#pragma unroll
    for (int i = 0; i < 16; ++i)
        b[i] = buf[(k0 + koff + i) * ldp + c0 + n];
    return b;
}

__device__ __forceinline__ v8f wmma_f16(v16h a, v16h b, v8f c) {
    return __builtin_amdgcn_wmma_f32_16x16x32_f16(
        /*neg_a=*/false, a, /*neg_b=*/false, b,
        /*c_mod=*/(short)0, c, /*reuse_a=*/false, /*reuse_b=*/false);
}

// Branch-free tanh: t = exp2(-2*log2(e)*|x|); tanh = copysign((1-t)/(1+t), x).
// Lowers to v_exp_f32 + v_rcp_f32 (TRANS, co-executes with VALU) + v_bfi_b32.
// Saturates correctly for large |x| (t -> 0 => +/-1). No divergent slow path.
__device__ __forceinline__ float fast_tanh(float v) {
    const float ax = __builtin_fabsf(v);
    const float t  = __builtin_amdgcn_exp2f(-2.8853900817779268f * ax);
    const float r  = (1.0f - t) * __builtin_amdgcn_rcpf(1.0f + t);
    return __builtin_copysignf(r, v);
}

__global__ __launch_bounds__(256)
void ctrnn_fused_kernel(const float* __restrict__ x,
                        const float* __restrict__ Wx,
                        const float* __restrict__ Wh,
                        const float* __restrict__ b,
                        const float* __restrict__ Wout,
                        const float* __restrict__ bout,
                        float* __restrict__ out) {
    __shared__ __align__(16) _Float16 wstage[H_ * LDW];  // 34816 B
    __shared__ __align__(16) _Float16 hbufA[16 * LDW];   //  4352 B (ping)
    __shared__ __align__(16) _Float16 hbufB[16 * LDW];   //  4352 B (pong)
    __shared__ __align__(16) _Float16 xbuf[16 * LDX];    //  2304 B

    const int tid  = threadIdx.x;
    const int lane = tid & 31;
    const int wave = tid >> 5;           // 0..7 -> N tile
    const int c0   = wave * 16;          // this wave's hidden-column base
    const int b0   = blockIdx.x * 16;    // batch-row base
    const int nIdx = lane & 15;
    const int half = lane >> 4;

    // ---- Stage Wh (f32 -> f16), gather per-wave B fragments, K=0..127 ----
    v16h WhB[4];
    for (int idx = tid; idx < H_ * H_; idx += 256)
        wstage[(idx >> 7) * LDW + (idx & 127)] = (_Float16)Wh[idx];
    __syncthreads();
#pragma unroll
    for (int k = 0; k < 4; ++k) WhB[k] = load_bfrag(wstage, LDW, lane, 32 * k, c0);
    __syncthreads();

    // ---- Stage Wx, gather B fragments, K=0..63 ----
    v16h WxB[2];
    for (int idx = tid; idx < F_ * H_; idx += 256)
        wstage[(idx >> 7) * LDW + (idx & 127)] = (_Float16)Wx[idx];
    __syncthreads();
#pragma unroll
    for (int k = 0; k < 2; ++k) WxB[k] = load_bfrag(wstage, LDW, lane, 32 * k, c0);
    __syncthreads();

    // ---- Stage Wout zero-padded to 16 columns, gather (used by wave 0) ----
    v16h WoB[4];
    for (int idx = tid; idx < H_ * 16; idx += 256) {
        const int kk = idx >> 4, cc = idx & 15;
        wstage[kk * LDW + cc] = (cc < O_) ? (_Float16)Wout[kk * O_ + cc]
                                          : (_Float16)0.0f;
    }
    __syncthreads();
#pragma unroll
    for (int k = 0; k < 4; ++k) WoB[k] = load_bfrag(wstage, LDW, lane, 32 * k, 0);

    // h0 = 0 (only the ping buffer is read first)
    for (int idx = tid; idx < 16 * LDW; idx += 256) hbufA[idx] = (_Float16)0.0f;
    __syncthreads();

    const float biasv = b[c0 + nIdx];
    const float boutv = (nIdx < O_) ? bout[nIdx] : 0.0f;

    v8f hreg = {};  // this wave's h columns; element r -> row r + 8*half

    for (int t = 0; t < T_; ++t) {
        // ---- load x tile [16 rows][64] f32 -> f16 LDS ----
        {
            const int row = tid >> 4;           // 0..15
            const int f   = (tid & 15) * 4;     // 0,4,...,60
            const float* src = x + ((size_t)(b0 + row) * T_ + t) * F_ + f;
            const v4f v = *(const v4f*)src;
            _Float16* dst = xbuf + row * LDX + f;
            dst[0] = (_Float16)v.x; dst[1] = (_Float16)v.y;
            dst[2] = (_Float16)v.z; dst[3] = (_Float16)v.w;
            if (t + 1 < T_) __builtin_prefetch(src + F_, 0, 0);  // next timestep
        }
        __syncthreads();

        // ---- xp = x_t @ Wx + b  (per-wave register C tile) ----
        v8f xp = {};
        {
            const v16h a0 = load_afrag(xbuf, LDX, lane, 0);
            const v16h a1 = load_afrag(xbuf, LDX, lane, 32);
            xp = wmma_f16(a0, WxB[0], xp);
            xp = wmma_f16(a1, WxB[1], xp);
#pragma unroll
            for (int r = 0; r < 8; ++r) xp[r] += biasv;
        }

        // ---- 6 explicit-Euler ODE substeps, ping-pong h tile ----
        // Substep u reads `cur`, writes `nxt`; one barrier per substep.
        // Each wave's gathers are consumed by its WMMAs (s_wait_dscnt) before
        // it signals the barrier, so reads of `cur` can never race the
        // following substep's writes to it.
#pragma unroll
        for (int u = 0; u < UNFOLD_; ++u) {
            const _Float16* cur = (u & 1) ? hbufB : hbufA;
            _Float16*       nxt = (u & 1) ? hbufA : hbufB;

            const v16h a0 = load_afrag(cur, LDW, lane, 0);
            const v16h a1 = load_afrag(cur, LDW, lane, 32);
            const v16h a2 = load_afrag(cur, LDW, lane, 64);
            const v16h a3 = load_afrag(cur, LDW, lane, 96);

            v8f acc = {};
            acc = wmma_f16(a0, WhB[0], acc);
            acc = wmma_f16(a1, WhB[1], acc);
            acc = wmma_f16(a2, WhB[2], acc);
            acc = wmma_f16(a3, WhB[3], acc);

#pragma unroll
            for (int r = 0; r < 8; ++r) {
                const float fval = fast_tanh(xp[r] + acc[r]);
                hreg[r] += DT_ * (fval - hreg[r]);  // h += DT*(-h/TAU + f), TAU=1
            }
            // publish updated columns for next substep's A gather
#pragma unroll
            for (int r = 0; r < 8; ++r)
                nxt[(r + 8 * half) * LDW + (c0 + nIdx)] = (_Float16)hreg[r];
            __syncthreads();
        }
        // UNFOLD_ is even -> final h is back in hbufA.

        // ---- logits_t = h @ Wout + bout  (wave 0 only; hbufA is stable) ----
        if (wave == 0) {
            const v16h a0 = load_afrag(hbufA, LDW, lane, 0);
            const v16h a1 = load_afrag(hbufA, LDW, lane, 32);
            const v16h a2 = load_afrag(hbufA, LDW, lane, 64);
            const v16h a3 = load_afrag(hbufA, LDW, lane, 96);
            v8f acc = {};
            acc = wmma_f16(a0, WoB[0], acc);
            acc = wmma_f16(a1, WoB[1], acc);
            acc = wmma_f16(a2, WoB[2], acc);
            acc = wmma_f16(a3, WoB[3], acc);
            if (nIdx < O_) {
#pragma unroll
                for (int r = 0; r < 8; ++r) {
                    const int row = r + 8 * half;
                    out[((size_t)(b0 + row) * T_ + t) * O_ + nIdx] = acc[r] + boutv;
                }
            }
        }
        // hbufA is next written in substep 1 of t+1, which is fenced behind
        // substep 0's barrier; wave 0's reads above are consumed before it
        // signals that barrier.
    }
}

extern "C" void kernel_launch(void* const* d_in, const int* in_sizes, int n_in,
                              void* d_out, int out_size, void* d_ws, size_t ws_size,
                              hipStream_t stream) {
    (void)in_sizes; (void)n_in; (void)d_ws; (void)ws_size; (void)out_size;
    const float* x    = (const float*)d_in[0];
    const float* Wx   = (const float*)d_in[1];
    const float* Wh   = (const float*)d_in[2];
    const float* b    = (const float*)d_in[3];
    const float* Wout = (const float*)d_in[4];
    const float* bout = (const float*)d_in[5];
    float* out = (float*)d_out;

    dim3 grid(B_ / 16);   // 16 workgroups, each owns 16 batch rows for all T
    dim3 block(256);      // 8 waves; wave w owns hidden columns [16w, 16w+16)
    ctrnn_fused_kernel<<<grid, block, 0, stream>>>(x, Wx, Wh, b, Wout, bout, out);
}
